// SelfConv_81922206204331
// MI455X (gfx1250) — matrix operand order repaired and checked by reading, not compiled, not run on hardware
//
#include <hip/hip_runtime.h>
#include <hip/hip_bf16.h>

// ---------------------------------------------------------------------------
// Problem constants (match reference):  B=8, C=96, H=256, W=256, NCHW f32
// ---------------------------------------------------------------------------
#define BATCH 8
#define CCH   96
#define HH    256
#define WWID  256
#define HWSZ  (HH * WWID)           // 65536 pixels per image
#define EPSV  1e-5f

typedef __attribute__((ext_vector_type(16))) __bf16        v16bf;
typedef __attribute__((ext_vector_type(8)))  float         v8f;
typedef __attribute__((ext_vector_type(4)))  unsigned int  v4u;
typedef __attribute__((ext_vector_type(8)))  int           v8i_t;
typedef __attribute__((ext_vector_type(4)))  int           v4i_t;

static __device__ __forceinline__ unsigned short f32_to_bf16(float f) {
    unsigned int u = __float_as_uint(f);
    unsigned int r = (u + 0x7FFFu + ((u >> 16) & 1u)) >> 16;  // round-nearest-even
    return (unsigned short)r;
}

// ---------------------------------------------------------------------------
// Kernel 1: per-pixel LayerNorm over channels (C=96), NCHW layout.
// One thread per pixel; 96 channel values held in registers (full unroll).
// ---------------------------------------------------------------------------
__global__ __launch_bounds__(256) void ln_kernel(
    const float* __restrict__ fea, const float* __restrict__ gamma,
    const float* __restrict__ beta, float* __restrict__ xln) {
    const int p  = blockIdx.x * 256 + threadIdx.x;   // 0 .. B*HW-1
    const int b  = p >> 16;                          // HW == 65536
    const int pp = p & 0xFFFF;

    const float* src = fea + (size_t)b * CCH * HWSZ + pp;
    float v[CCH];
    float s = 0.f;
#pragma unroll
    for (int c = 0; c < CCH; ++c) {
        v[c] = src[(size_t)c * HWSZ];
        s += v[c];
    }
    const float mu = s * (1.0f / CCH);
    float q = 0.f;
#pragma unroll
    for (int c = 0; c < CCH; ++c) {
        const float d = v[c] - mu;
        q += d * d;
    }
    const float inv = rsqrtf(q * (1.0f / CCH) + EPSV);

    float* dst = xln + (size_t)b * CCH * HWSZ + pp;
#pragma unroll
    for (int c = 0; c < CCH; ++c) {
        dst[(size_t)c * HWSZ] = (v[c] - mu) * inv * gamma[c] + beta[c];
    }
}

// ---------------------------------------------------------------------------
// Kernel 2: fused depthwise 3x3 -> 1x5 -> 5x1 (+biases), per-channel tiles.
// Block = 4 channels x 32x32 output tile, 3-wide halo.
// Interior tiles: halo tile loaded by the Tensor Data Mover (one
// tensor_load_to_lds per channel, issued by wave 0; TDM ignores EXEC), then
// s_wait_tensorcnt + barrier. Border tiles: manual zero-padded loads.
// Output y3 stored channel-pair packed:  dword = {bf16(ci+1), bf16(ci)} at
// y3p[(b*48 + ci/2)*HW + p]  -> WMMA B fragments become single b32 loads.
// ---------------------------------------------------------------------------
#define CG  4
#define TS  32
#define XT  38          // TS + 2*3 halo
#define Y1T 36          // TS + 2*2

__global__ __launch_bounds__(256) void dw_kernel(
    const float* __restrict__ xln,
    const float* __restrict__ w1, const float* __restrict__ b1,
    const float* __restrict__ w2, const float* __restrict__ b2,
    const float* __restrict__ w3, const float* __restrict__ b3,
    unsigned int* __restrict__ y3p) {
    __shared__ float xs [CG][XT ][XT];   // x_ln tile (TDM target; reused for y2)
    __shared__ float y1s[CG][Y1T][40];   // after 3x3 (padded pitch)
    __shared__ float wl1[CG][9], wl2[CG][5], wl3[CG][5];
    __shared__ float bl1[CG], bl2[CG], bl3[CG];

    int t = blockIdx.x;
    const int tx = t & 7;  t >>= 3;          // W/32 = 8
    const int ty = t & 7;  t >>= 3;          // H/32 = 8
    const int cg = t % (CCH / CG);
    const int b  = t / (CCH / CG);
    const int c0 = cg * CG;
    const int row0 = ty * TS, col0 = tx * TS;
    const int tid = threadIdx.x;

    // Stage weights/biases for this channel group.
    if (tid < CG * 9) {
        wl1[tid / 9][tid % 9] = w1[(c0 + tid / 9) * 9 + tid % 9];
    } else if (tid < CG * 9 + CG * 5) {
        const int i = tid - CG * 9;
        wl2[i / 5][i % 5] = w2[(c0 + i / 5) * 5 + i % 5];
    } else if (tid < CG * 9 + 2 * CG * 5) {
        const int i = tid - CG * 9 - CG * 5;
        wl3[i / 5][i % 5] = w3[(c0 + i / 5) * 5 + i % 5];
    } else if (tid < CG * 9 + 2 * CG * 5 + CG) {
        const int i = tid - CG * 9 - 2 * CG * 5;
        bl1[i] = b1[c0 + i];
    } else if (tid < CG * 9 + 2 * CG * 5 + 2 * CG) {
        const int i = tid - CG * 9 - 2 * CG * 5 - CG;
        bl2[i] = b2[c0 + i];
    } else if (tid < CG * 9 + 2 * CG * 5 + 3 * CG) {
        const int i = tid - CG * 9 - 2 * CG * 5 - 2 * CG;
        bl3[i] = b3[c0 + i];
    }

    const bool interior = (tx >= 1) && (tx <= 6) && (ty >= 1) && (ty <= 6);

    if (interior) {
        // ---- TDM path: 2D tile DMA, one descriptor per channel ----
        if (tid < 32) {                       // wave 0 only (TDM ignores EXEC)
#pragma unroll
            for (int ch = 0; ch < CG; ++ch) {
                const unsigned long long gaddr = (unsigned long long)(const char*)(
                    xln + ((size_t)b * CCH + (c0 + ch)) * HWSZ +
                    (size_t)(row0 - 3) * WWID + (col0 - 3));
                const unsigned lds_addr =
                    (unsigned)(unsigned long long)(void*)&xs[ch][0][0];
                // D# group0: count=1 | lds_addr | global_addr | type=2
                v4u g0;
                g0[0] = 1u;
                g0[1] = lds_addr;
                g0[2] = (unsigned)gaddr;
                g0[3] = (unsigned)(gaddr >> 32) | (2u << 30);
                // D# group1 (bitfields per CDNA5 ISA):
                v8i_t g1;
                g1[0] = 0x00020000;            // wg_mask=0, data_size=4B
                g1[1] = (int)(38u << 16);      // tensor_dim0 = 38 (lo16)
                g1[2] = (int)(38u << 16);      // tensor_dim1 = 38 (lo16)
                g1[3] = (int)(38u << 16);      // tile_dim0  = 38
                g1[4] = 38;                    // tile_dim1 = 38, tile_dim2 = 0
                g1[5] = WWID;                  // tensor_dim0_stride = 256
                g1[6] = 0;
                g1[7] = 0;
                const v4i_t z4 = {0, 0, 0, 0};
                const v8i_t z8 = {0, 0, 0, 0, 0, 0, 0, 0};
                __builtin_amdgcn_tensor_load_to_lds(g0, g1, z4, z4, z8, 0);
            }
            __builtin_amdgcn_s_wait_tensorcnt(0);
        }
    } else {
        // ---- Border tiles: manual zero-padded load ----
        for (int idx = tid; idx < CG * XT * XT; idx += 256) {
            const int ch  = idx / (XT * XT);
            const int rem = idx % (XT * XT);
            const int r = rem / XT, c = rem % XT;
            const int gr = row0 + r - 3, gc = col0 + c - 3;
            float v = 0.f;
            if (gr >= 0 && gr < HH && gc >= 0 && gc < WWID)
                v = xln[((size_t)b * CCH + (c0 + ch)) * HWSZ + (size_t)gr * WWID + gc];
            xs[ch][r][c] = v;
        }
    }
    __syncthreads();

    // dw1: 3x3 over xs -> y1s (36x36 region, x-coords 1..36).
    for (int idx = tid; idx < CG * Y1T * Y1T; idx += 256) {
        const int ch  = idx / (Y1T * Y1T);
        const int rem = idx % (Y1T * Y1T);
        const int r = rem / Y1T, c = rem % Y1T;
        float acc = bl1[ch];
#pragma unroll
        for (int di = 0; di < 3; ++di)
#pragma unroll
            for (int dj = 0; dj < 3; ++dj)
                acc += wl1[ch][di * 3 + dj] * xs[ch][r + di][c + dj];
        y1s[ch][r][c] = acc;
    }
    __syncthreads();

    // dw2: 1x5 horizontal over y1s -> y2 (36 rows x 32 cols), reuse xs.
    for (int idx = tid; idx < CG * Y1T * TS; idx += 256) {
        const int ch  = idx / (Y1T * TS);
        const int rem = idx % (Y1T * TS);
        const int r = rem / TS, c = rem % TS;
        float acc = bl2[ch];
#pragma unroll
        for (int k = 0; k < 5; ++k)
            acc += wl2[ch][k] * y1s[ch][r][c + k];
        xs[ch][r][c] = acc;          // xs is dead -> holds y2 now
    }
    __syncthreads();

    // dw3: 5x1 vertical over y2 -> y3 (32x32); both channels of a pair
    // computed by one thread, stored as one packed bf16x2 dword (coalesced).
    for (int idx = tid; idx < (CG / 2) * TS * TS; idx += 256) {
        const int chp = idx / (TS * TS);
        const int rem = idx % (TS * TS);
        const int r = rem / TS, c = rem % TS;
        const int ch0 = 2 * chp, ch1 = ch0 + 1;
        float a0 = bl3[ch0], a1 = bl3[ch1];
#pragma unroll
        for (int k = 0; k < 5; ++k) {
            a0 += wl3[ch0][k] * xs[ch0][r + k][c];
            a1 += wl3[ch1][k] * xs[ch1][r + k][c];
        }
        const unsigned pk =
            (unsigned)f32_to_bf16(a0) | ((unsigned)f32_to_bf16(a1) << 16);
        y3p[((size_t)b * (CCH / 2) + (c0 >> 1) + chp) * HWSZ +
            (size_t)(row0 + r) * WWID + (col0 + c)] = pk;
    }
}

// ---------------------------------------------------------------------------
// Kernel 3: pointwise 96x96 GEMM via v_wmma_f32_16x16x32_bf16 + gate.
//   out[b,co,p] = (sum_ci W[co,ci] * y3[b,ci,p]) * xln[b,co,p]
// Wave: 48 output channels (3 M-blocks) x 16 pixels x 4 chunk iterations.
// A fragments built once from LDS-staged bf16 W; B fragment VGPRs are single
// b32 loads from the channel-pair-packed y3 (already in WMMA lane format).
// ---------------------------------------------------------------------------
__global__ __launch_bounds__(256) void pw_kernel(
    const unsigned int* __restrict__ y3p, const float* __restrict__ pw,
    const float* __restrict__ xln, float* __restrict__ out) {
    __shared__ unsigned int sW[96 * 48];   // bf16 pairs, row-major [co][ci/2]

    const int tid = threadIdx.x;
    const int b       = blockIdx.x >> 8;             // 256 blocks per image
    const int pixBase = (blockIdx.x & 255) * 256;    // 256 pixels per block

    // Stage pw_w as packed bf16 (two consecutive ci per dword).
    for (int i = tid; i < 96 * 48; i += 256) {
        const int co = i / 48, cp = i % 48;
        const float f0 = pw[co * 96 + 2 * cp];
        const float f1 = pw[co * 96 + 2 * cp + 1];
        sW[i] = (unsigned int)f32_to_bf16(f0) | ((unsigned int)f32_to_bf16(f1) << 16);
    }
    __syncthreads();

    const int lane = tid & 31;
    const int wave = tid >> 5;
    const int hlf  = (lane >> 4) & 1;     // lane half (0: lanes 0-15, 1: 16-31)
    const int n    = lane & 15;
    const int co0  = (wave >> 2) * 48;    // waves 0-3 -> co 0..47, 4-7 -> 48..95
    const int pw0  = pixBase + (wave & 3) * 16;

    union Frag { unsigned int u[8]; v16bf v; };

    // A fragments: 3 M-blocks x 3 K-chunks. 16-bit A 16x32 layout:
    //   lane half 0: V0..V3 -> K=0..7 (pairs), V4..V7 -> K=16..23
    //   lane half 1: V0..V3 -> K=8..15,        V4..V7 -> K=24..31
    Frag a[3][3];
#pragma unroll
    for (int mb = 0; mb < 3; ++mb) {
        const int row = co0 + mb * 16 + n;           // M = lane & 15
#pragma unroll
        for (int kc = 0; kc < 3; ++kc) {
#pragma unroll
            for (int r = 0; r < 8; ++r) {
                const int kb = ((r & 4) << 2) + hlf * 8 + 2 * (r & 3);
                a[mb][kc].u[r] = sW[row * 48 + ((kc * 32 + kb) >> 1)];
            }
        }
    }

    const size_t imgBase  = (size_t)b * CCH * HWSZ;
    const size_t imgBaseP = (size_t)b * (CCH / 2) * HWSZ;

    for (int it = 0; it < 4; ++it) {
        const int p0 = pw0 + it * 64;

        // Prefetch the gate lines needed in the epilogue.
#pragma unroll
        for (int mb = 0; mb < 3; ++mb)
            __builtin_prefetch(&xln[imgBase + (size_t)(co0 + mb * 16) * HWSZ + p0], 0, 3);

        // B fragments: 16-bit B 32x16 layout (same K order as A); VGPR r of chunk kc
        // holds K = kc*32 + (r&4)*4 + hlf*8 + 2*(r&3) + {0,1} for this lane's column n.
        // Packed y3 dword at [b][kc*16 + (r&4)*2 + hlf*4 + (r&3)][p0+n] is exactly that pair.
        Frag bf[3];
#pragma unroll
        for (int kc = 0; kc < 3; ++kc) {
#pragma unroll
            for (int r = 0; r < 8; ++r) {
                const int kk = kc * 16 + (r & 4) * 2 + hlf * 4 + (r & 3);
                bf[kc].u[r] = y3p[imgBaseP + (size_t)kk * HWSZ + p0 + n];
            }
        }

        v8f acc[3] = {};   // 16x16 f32 accumulators, zero C
#pragma unroll
        for (int kc = 0; kc < 3; ++kc) {
#pragma unroll
            for (int mb = 0; mb < 3; ++mb) {
                acc[mb] = __builtin_amdgcn_wmma_f32_16x16x32_bf16(
                    /*neg_a=*/false, a[mb][kc].v, /*neg_b=*/false, bf[kc].v,
                    /*c_mod=*/(short)0, acc[mb], /*reuse_a=*/false, /*reuse_b=*/false);
            }
        }

        // Epilogue: D layout (f32 16x16): VGPR r -> M = r + 8*half, N = lane&15.
#pragma unroll
        for (int mb = 0; mb < 3; ++mb) {
#pragma unroll
            for (int r = 0; r < 8; ++r) {
                const int co = co0 + mb * 16 + r + hlf * 8;
                const size_t g = imgBase + (size_t)co * HWSZ + p0 + n;
                out[g] = acc[mb][r] * xln[g];
            }
        }
    }
}

// ---------------------------------------------------------------------------
// Host launcher. Workspace layout:
//   [0, 201326592)           x_ln f32   (B*C*H*W*4)
//   [201326592, +100663296)  y3  packed bf16x2 (B*(C/2)*H*W*4)
// ---------------------------------------------------------------------------
extern "C" void kernel_launch(void* const* d_in, const int* in_sizes, int n_in,
                              void* d_out, int out_size, void* d_ws, size_t ws_size,
                              hipStream_t stream) {
    const float* fea = (const float*)d_in[0];
    const float* g   = (const float*)d_in[1];
    const float* be  = (const float*)d_in[2];
    const float* w1  = (const float*)d_in[3];
    const float* b1  = (const float*)d_in[4];
    const float* w2  = (const float*)d_in[5];
    const float* b2  = (const float*)d_in[6];
    const float* w3  = (const float*)d_in[7];
    const float* b3  = (const float*)d_in[8];
    const float* pw  = (const float*)d_in[9];
    float* out = (float*)d_out;

    float*        xln = (float*)d_ws;
    unsigned int* y3p = (unsigned int*)((char*)d_ws + (size_t)BATCH * CCH * HWSZ * 4);

    ln_kernel<<<BATCH * HWSZ / 256, 256, 0, stream>>>(fea, g, be, xln);
    dw_kernel<<<BATCH * (CCH / CG) * (HH / TS) * (WWID / TS), 256, 0, stream>>>(
        xln, w1, b1, w2, b2, w3, b3, y3p);
    pw_kernel<<<BATCH * (HWSZ / 256), 256, 0, stream>>>(y3p, pw, xln, out);
}
